// LSTM_28346784153891
// MI455X (gfx1250) — matrix operand order repaired and checked by reading, not compiled, not run on hardware
//
#include <hip/hip_runtime.h>
#include <hip/hip_bf16.h>

typedef __bf16 bf16_t;
typedef __attribute__((ext_vector_type(16))) __bf16 v16bf;
typedef __attribute__((ext_vector_type(8)))  __bf16 v8bf;
typedef __attribute__((ext_vector_type(8)))  float  v8f;

#define LSTM_B   64
#define LSTM_T   256
#define LSTM_NIN 512
#define LSTM_H   1024
#define LSTM_G   4096   // 4*H

// ---------------------------------------------------------------------------
// WMMA fragment loaders (v_wmma_f32_16x16x32_bf16, wave32)
// A: 16x32 (MxK).  lanes 0-15: M=lane, K[0:8)+K[16:24); lanes 16-31: M=lane-16,
//    K[8:16)+K[24:32)  (packed pairs per VGPR per ISA 7.12.2)
// B: 32x16 (KxN).  lane 0-15: col N=lane, K[0:8)+K[16:24); lane 16-31: col N=lane-16, K[8:16)+K[24:32)
// C: v8f, VGPR r -> M = r (lanes 0-15) / 8+r (lanes 16-31), N = lane&15
// ---------------------------------------------------------------------------
__device__ __forceinline__ v16bf load_fragA(const bf16_t* __restrict__ base, int ld, int lane) {
    const int m  = lane & 15;
    const int kh = (lane >> 4) << 3;          // 0 or 8
    const bf16_t* p = base + (size_t)m * ld + kh;
    union { v16bf v; v8bf h[2]; } u;
    u.h[0] = *(const v8bf*)(p);
    u.h[1] = *(const v8bf*)(p + 16);
    return u.v;
}

__device__ __forceinline__ v16bf load_fragB(const bf16_t* __restrict__ base, int ld, int lane) {
    const int n  = lane & 15;
    const int ks = (lane >> 4) << 3;          // 0 or 8
    const bf16_t* p = base + (size_t)n * ld + ks;
    union { v16bf v; v8bf h[2]; } u;
    u.h[0] = *(const v8bf*)(p);
    u.h[1] = *(const v8bf*)(p + 16);
    return u.v;
}

__device__ __forceinline__ v8f wmma_bf16(v16bf a, v16bf b, v8f c) {
    return __builtin_amdgcn_wmma_f32_16x16x32_bf16(false, a, false, b, (short)0, c, false, false);
}

// ---------------------------------------------------------------------------
// f32 -> bf16 conversion
// ---------------------------------------------------------------------------
__global__ void cvt_bf16_kernel(const float* __restrict__ s, bf16_t* __restrict__ d, long n) {
    long i = (long)blockIdx.x * blockDim.x + threadIdx.x;
    if (i < n) d[i] = (bf16_t)s[i];
}

// ---------------------------------------------------------------------------
// XW = X[M,K](bf16) * W[4096,K]^T(bf16) + b1 + b2   ->  f32 out
// REMAP=1: X rows are b*T+t (layer-0 input) -> out row t*B+b; REMAP=0: identity.
// Block: 256 thr (8 waves), tile 64(M) x 128(N); wave: 16x64 (4 accumulators).
// Ping-pong double buffer, fully unrolled; sched_barrier(0) after each load
// group pins the software pipeline so chunk k+1 loads stay above chunk k's
// WMMA group (partial loadcnt waits instead of full drains).
// ---------------------------------------------------------------------------
template <int K, int REMAP>
__global__ void __launch_bounds__(256)
gemm_xw_kernel(const bf16_t* __restrict__ X, const bf16_t* __restrict__ W,
               const float* __restrict__ b1, const float* __restrict__ b2,
               float* __restrict__ out)
{
    const int lane = threadIdx.x & 31;
    const int w    = threadIdx.x >> 5;                // 0..7
    const int m0   = blockIdx.x * 64  + (w & 3) * 16;
    const int n0   = blockIdx.y * 128 + (w >> 2) * 64;

    v8f zero = {};
    v8f acc[4] = {zero, zero, zero, zero};

    const bf16_t* xrow = X + (size_t)m0 * K;
    const bf16_t* wb[4];
#pragma unroll
    for (int i = 0; i < 4; ++i) wb[i] = W + (size_t)(n0 + i * 16) * K;

    v16bf aA, aB, bA[4], bB[4];
    auto load_tile = [&](int k, v16bf& a, v16bf (&b)[4]) {
        a = load_fragA(xrow + k, K, lane);
#pragma unroll
        for (int i = 0; i < 4; ++i) b[i] = load_fragB(wb[i] + k, K, lane);
    };
    auto compute = [&](v16bf& a, v16bf (&b)[4]) {
#pragma unroll
        for (int i = 0; i < 4; ++i) acc[i] = wmma_bf16(a, b[i], acc[i]);
    };

    load_tile(0, aA, bA);
    load_tile(32, aB, bB);
    __builtin_amdgcn_sched_barrier(0);
#pragma unroll
    for (int k = 64; k < K; k += 64) {
        compute(aA, bA);            // consume tile k-64 (loads for k-32 in flight)
        load_tile(k, aA, bA);       // refill A with tile k
        __builtin_amdgcn_sched_barrier(0);
        compute(aB, bB);            // consume tile k-32 (loads for k in flight)
        load_tile(k + 32, aB, bB);
        __builtin_amdgcn_sched_barrier(0);
    }
    compute(aA, bA);
    compute(aB, bB);

    const int mbase = m0 + ((lane >> 4) << 3);
#pragma unroll
    for (int i = 0; i < 4; ++i) {
        const int n = n0 + i * 16 + (lane & 15);
        const float bias = b1[n] + b2[n];
#pragma unroll
        for (int r = 0; r < 8; ++r) {
            const int m   = mbase + r;
            const int row = REMAP ? ((m % LSTM_T) * LSTM_B + (m / LSTM_T)) : m;
            out[(size_t)row * LSTM_G + n] = acc[i][r] + bias;
        }
    }
}

// ---------------------------------------------------------------------------
// One recurrent step:  gates = xw_t + h_prev @ w_hh^T ; LSTM cell update.
// Grid: H/32 = 32 blocks; block 256 thr (8 waves).  Each wave owns a
// 16(batch) x 16(hid) tile and its four gate accumulators (gate g lives at
// column n + g*H), so the cell nonlinearity fuses on the WMMA outputs.
// Same pinned ping-pong pipeline as the GEMM (K = H = 1024).
// ---------------------------------------------------------------------------
__global__ void __launch_bounds__(256)
lstm_step_kernel(const bf16_t* __restrict__ h_prev,   // [B,H] bf16
                 const float*  __restrict__ xw_t,     // [B,4H] f32 (bias included)
                 const bf16_t* __restrict__ w_hh,     // [4H,H] bf16
                 float*        __restrict__ c_state,  // [B,H] f32 (in/out)
                 bf16_t*       __restrict__ h_out,    // [B,H] bf16 (chain)
                 float*        __restrict__ h_f32)    // [B,H] f32 (final-state mirror)
{
    const int lane = threadIdx.x & 31;
    const int w    = threadIdx.x >> 5;                // 0..7
    const int m16  = (w & 3) * 16;                    // batch tile (B=64)
    const int nb   = blockIdx.x * 32 + (w >> 2) * 16; // hidden column base

    v8f zero = {};
    v8f acc[4] = {zero, zero, zero, zero};

    const bf16_t* hrow = h_prev + (size_t)m16 * LSTM_H;
    const bf16_t* wb[4];
#pragma unroll
    for (int g = 0; g < 4; ++g) wb[g] = w_hh + (size_t)(g * LSTM_H + nb) * LSTM_H;

    v16bf aA, aB, bA[4], bB[4];
    auto load_tile = [&](int k, v16bf& a, v16bf (&b)[4]) {
        a = load_fragA(hrow + k, LSTM_H, lane);
#pragma unroll
        for (int g = 0; g < 4; ++g) b[g] = load_fragB(wb[g] + k, LSTM_H, lane);
    };
    auto compute = [&](v16bf& a, v16bf (&b)[4]) {
#pragma unroll
        for (int g = 0; g < 4; ++g) acc[g] = wmma_bf16(a, b[g], acc[g]);
    };

    load_tile(0, aA, bA);
    load_tile(32, aB, bB);
    __builtin_amdgcn_sched_barrier(0);
#pragma unroll
    for (int k = 64; k < LSTM_H; k += 64) {
        compute(aA, bA);
        load_tile(k, aA, bA);
        __builtin_amdgcn_sched_barrier(0);
        compute(aB, bB);
        load_tile(k + 32, aB, bB);
        __builtin_amdgcn_sched_barrier(0);
    }
    compute(aA, bA);
    compute(aB, bB);

    const int mbase = m16 + ((lane >> 4) << 3);
    const int n     = nb + (lane & 15);
#pragma unroll
    for (int r = 0; r < 8; ++r) {
        const int m = mbase + r;
        const size_t gx = (size_t)m * LSTM_G;
        const float xi = xw_t[gx + 0 * LSTM_H + n] + acc[0][r];
        const float xf = xw_t[gx + 1 * LSTM_H + n] + acc[1][r];
        const float xg = xw_t[gx + 2 * LSTM_H + n] + acc[2][r];
        const float xo = xw_t[gx + 3 * LSTM_H + n] + acc[3][r];
        const float ig = 1.f / (1.f + __expf(-xi));
        const float fg = 1.f / (1.f + __expf(-xf));
        const float gg = tanhf(xg);
        const float og = 1.f / (1.f + __expf(-xo));
        const size_t hx = (size_t)m * LSTM_H + n;
        const float cc = fg * c_state[hx] + ig * gg;
        c_state[hx] = cc;
        const float hh = og * tanhf(cc);
        h_out[hx] = (bf16_t)hh;
        h_f32[hx] = hh;
    }
}

// ---------------------------------------------------------------------------
// decoded = hT @ w_dec^T + b_dec ; log_softmax.  One block per batch row.
// ---------------------------------------------------------------------------
__global__ void __launch_bounds__(256)
decoder_kernel(const float* __restrict__ hT,     // [B,H]
               const float* __restrict__ w_dec,  // [NIN,H]
               const float* __restrict__ b_dec,  // [NIN]
               float* __restrict__ logp)         // [B,NIN]
{
    __shared__ float hrow[LSTM_H];
    __shared__ float logits[LSTM_NIN];
    __shared__ float red[256];
    const int b = blockIdx.x, tid = threadIdx.x;

    for (int i = tid; i < LSTM_H; i += 256) hrow[i] = hT[(size_t)b * LSTM_H + i];
    __syncthreads();

    for (int o = tid; o < LSTM_NIN; o += 256) {
        const float* wr = w_dec + (size_t)o * LSTM_H;
        float s = b_dec[o];
        for (int k = 0; k < LSTM_H; ++k) s += hrow[k] * wr[k];
        logits[o] = s;
    }
    __syncthreads();

    float m = -INFINITY;
    for (int o = tid; o < LSTM_NIN; o += 256) m = fmaxf(m, logits[o]);
    red[tid] = m; __syncthreads();
    for (int s = 128; s > 0; s >>= 1) { if (tid < s) red[tid] = fmaxf(red[tid], red[tid + s]); __syncthreads(); }
    const float mx = red[0]; __syncthreads();

    float sum = 0.f;
    for (int o = tid; o < LSTM_NIN; o += 256) sum += __expf(logits[o] - mx);
    red[tid] = sum; __syncthreads();
    for (int s = 128; s > 0; s >>= 1) { if (tid < s) red[tid] += red[tid + s]; __syncthreads(); }
    const float lse = mx + __logf(red[0]);
    __syncthreads();

    for (int o = tid; o < LSTM_NIN; o += 256)
        logp[(size_t)b * LSTM_NIN + o] = logits[o] - lse;
}

// ---------------------------------------------------------------------------
extern "C" void kernel_launch(void* const* d_in, const int* in_sizes, int n_in,
                              void* d_out, int out_size, void* d_ws, size_t ws_size,
                              hipStream_t stream) {
    const int B = LSTM_B, T = LSTM_T, NIN = LSTM_NIN, H = LSTM_H, G = LSTM_G;
    const size_t BT = (size_t)B * T;
    const size_t BH = (size_t)B * H;

    const float* x      = (const float*)d_in[0];
    const float* h0     = (const float*)d_in[1];
    const float* c0     = (const float*)d_in[2];
    const float* w_ih_0 = (const float*)d_in[3];
    const float* w_hh_0 = (const float*)d_in[4];
    const float* b_ih_0 = (const float*)d_in[5];
    const float* b_hh_0 = (const float*)d_in[6];
    const float* w_ih_1 = (const float*)d_in[7];
    const float* w_hh_1 = (const float*)d_in[8];
    const float* b_ih_1 = (const float*)d_in[9];
    const float* b_hh_1 = (const float*)d_in[10];
    const float* w_dec  = (const float*)d_in[11];
    const float* b_dec  = (const float*)d_in[12];
    float* out = (float*)d_out;   // [0,32768): logp  [32768,163840): hn  [163840,294912): cn

    // ---- workspace carve-out (256B aligned) ----
    char* ws = (char*)d_ws;
    size_t off = 0;
    auto alloc = [&](size_t bytes) -> char* {
        char* p = ws + off;
        off = (off + bytes + 255) & ~(size_t)255;
        return p;
    };
    bf16_t* xbf   = (bf16_t*)alloc(BT * NIN * sizeof(bf16_t));
    bf16_t* wih0  = (bf16_t*)alloc((size_t)G * NIN * sizeof(bf16_t));
    bf16_t* whh0  = (bf16_t*)alloc((size_t)G * H * sizeof(bf16_t));
    bf16_t* wih1  = (bf16_t*)alloc((size_t)G * H * sizeof(bf16_t));
    bf16_t* whh1  = (bf16_t*)alloc((size_t)G * H * sizeof(bf16_t));
    float*  xw    = (float*)alloc(BT * G * sizeof(float));      // [T][B][4H], reused per layer
    bf16_t* chain = (bf16_t*)alloc(BT * H * sizeof(bf16_t));    // [T][B][H], reused per layer
    bf16_t* hinit = (bf16_t*)alloc(BH * sizeof(bf16_t));
    float*  cst0  = (float*)alloc(BH * sizeof(float));
    float*  cst1  = (float*)alloc(BH * sizeof(float));
    float*  hf0   = (float*)alloc(BH * sizeof(float));
    float*  hf1   = (float*)alloc(BH * sizeof(float));

    auto cvt = [&](const float* s, bf16_t* d, long n) {
        cvt_bf16_kernel<<<dim3((unsigned)((n + 255) / 256)), dim3(256), 0, stream>>>(s, d, n);
    };

    // ---- stage bf16 operands (weights stay L2-resident for all 512 steps) ----
    cvt(x,      xbf,  (long)BT * NIN);
    cvt(w_ih_0, wih0, (long)G * NIN);
    cvt(w_hh_0, whh0, (long)G * H);
    cvt(w_ih_1, wih1, (long)G * H);
    cvt(w_hh_1, whh1, (long)G * H);

    // ================= layer 0 =================
    gemm_xw_kernel<LSTM_NIN, 1><<<dim3((unsigned)(BT / 64), G / 128), dim3(256), 0, stream>>>(
        xbf, wih0, b_ih_0, b_hh_0, xw);

    cvt(h0, hinit, (long)BH);
    hipMemcpyAsync(cst0, c0, BH * sizeof(float), hipMemcpyDeviceToDevice, stream);

    for (int t = 0; t < T; ++t) {
        const bf16_t* hp = (t == 0) ? hinit : chain + (size_t)(t - 1) * BH;
        lstm_step_kernel<<<dim3(H / 32), dim3(256), 0, stream>>>(
            hp, xw + (size_t)t * B * G, whh0, cst0, chain + (size_t)t * BH, hf0);
    }

    // ================= layer 1 =================
    gemm_xw_kernel<LSTM_H, 0><<<dim3((unsigned)(BT / 64), G / 128), dim3(256), 0, stream>>>(
        chain, wih1, b_ih_1, b_hh_1, xw);

    cvt(h0 + BH, hinit, (long)BH);
    hipMemcpyAsync(cst1, c0 + BH, BH * sizeof(float), hipMemcpyDeviceToDevice, stream);

    for (int t = 0; t < T; ++t) {
        const bf16_t* hp = (t == 0) ? hinit : chain + (size_t)(t - 1) * BH;
        lstm_step_kernel<<<dim3(H / 32), dim3(256), 0, stream>>>(
            hp, xw + (size_t)t * B * G, whh1, cst1, chain + (size_t)t * BH, hf1);
    }

    // ================= decode + pack outputs =================
    decoder_kernel<<<dim3(B), dim3(256), 0, stream>>>(hf1, w_dec, b_dec, out);

    float* hn = out + (size_t)B * NIN;          // 32768
    float* cn = hn + 2 * BH;                    // 163840
    hipMemcpyAsync(hn,      hf0,  BH * sizeof(float), hipMemcpyDeviceToDevice, stream);
    hipMemcpyAsync(hn + BH, hf1,  BH * sizeof(float), hipMemcpyDeviceToDevice, stream);
    hipMemcpyAsync(cn,      cst0, BH * sizeof(float), hipMemcpyDeviceToDevice, stream);
    hipMemcpyAsync(cn + BH, cst1, BH * sizeof(float), hipMemcpyDeviceToDevice, stream);
}
